// DPWA_80771154968936
// MI455X (gfx1250) — matrix operand order repaired and checked by reading, not compiled, not run on hardware
//
#include <hip/hip_runtime.h>
#include <hip/hip_bf16.h>

// ---------------------------------------------------------------------------
// MI455X (gfx1250) dual-partition window attention, bf16 WMMA pipeline.
// GEMMs + attention through v_wmma_f32_16x16x32_bf16 (f32 accumulate);
// global->LDS staging via GLOBAL_LOAD_ASYNC_TO_LDS_B128 (ASYNCcnt).
// ---------------------------------------------------------------------------

typedef __bf16 bf16;
typedef __attribute__((ext_vector_type(16))) __bf16 v16bf;
typedef __attribute__((ext_vector_type(8)))  __bf16 v8bf;
typedef __attribute__((ext_vector_type(8)))  float  v8f;

#define WMMA_BF16(a,b,c) \
  __builtin_amdgcn_wmma_f32_16x16x32_bf16(false,(a),false,(b),(short)0,(c),false,false)

constexpr int Bn   = 8;
constexpr int Cc   = 384;
constexpr int HW   = 4096;      // 64*64
constexpr int Mtok = Bn * HW;   // 32768
constexpr int NN3  = 3 * Cc;    // 1152
constexpr int CH   = 192;       // channels per branch
constexpr int HD   = 48;        // head dim
constexpr int SS_  = 4;
constexpr float SCALE_ = 0.14433756729740643f; // 48^-0.5

// --- CDNA5 async global->LDS copy (16 bytes/lane), ASYNCcnt tracked --------
__device__ __forceinline__ void async_copy16(void* lds_ptr, const void* gptr) {
  uint32_t lds_off = (uint32_t)(uintptr_t)lds_ptr;   // low 32 bits = LDS offset
  asm volatile("global_load_async_to_lds_b128 %0, %1, off"
               :: "v"(lds_off), "v"((uint64_t)(uintptr_t)gptr)
               : "memory");
}
__device__ __forceinline__ void wait_async0() {
  asm volatile("s_wait_asynccnt 0x0" ::: "memory");
}

// --- WMMA fragment loaders from LDS ---------------------------------------
// A (16x32 bf16): M = lane%16; lane<16 holds K {0..7,16..23}, lane>=16 {8..15,24..31}
__device__ __forceinline__ v16bf frag_a(const bf16* base, int ld, int row, int koff, int lane) {
  const bf16* p = base + row * ld + koff + ((lane & 16) ? 8 : 0);
  v8bf lo = *(const v8bf*)p;
  v8bf hi = *(const v8bf*)(p + 16);
  v16bf r;
#pragma unroll
  for (int i = 0; i < 8; ++i) { r[i] = lo[i]; r[i + 8] = hi[i]; }
  return r;
}
// B (32x16 bf16), stored column-per-row in LDS: N = lane%16; lane<16 holds K {0..7,16..23},
// lane>=16 holds K {8..15,24..31}.
__device__ __forceinline__ v16bf frag_b(const bf16* base, int ld, int col, int koff, int lane) {
  const bf16* p = base + col * ld + koff + ((lane & 16) ? 8 : 0);
  v8bf lo = *(const v8bf*)p;
  v8bf hi = *(const v8bf*)(p + 16);
  v16bf r;
#pragma unroll
  for (int i = 0; i < 8; ++i) { r[i] = lo[i]; r[i + 8] = hi[i]; }
  return r;
}

// --- weights fp32 -> bf16 --------------------------------------------------
__global__ void k_cvt(const float* __restrict__ qkv_w, const float* __restrict__ proj_w,
                      bf16* __restrict__ wq, bf16* __restrict__ wp) {
  int i = blockIdx.x * 256 + threadIdx.x;
  const int n1 = Cc * NN3, n2 = Cc * Cc;
  if (i < n1)            wq[i]      = (bf16)qkv_w[i];
  else if (i < n1 + n2)  wp[i - n1] = (bf16)proj_w[i - n1];
}

// --- LayerNorm over channels, emit bf16 token-major activations ------------
__global__ __launch_bounds__(128) void k_ln(const float* __restrict__ x,
                                            const float* __restrict__ g,
                                            const float* __restrict__ bta,
                                            bf16* __restrict__ Xn) {
  int m = blockIdx.x * 128 + threadIdx.x;        // token id
  int b = m >> 12, p = m & 4095;
  const float* xp = x + (size_t)b * Cc * HW + p; // x[b][c][p], stride HW over c
  float s = 0.f, s2 = 0.f;
  for (int c = 0; c < Cc; ++c) { float v = xp[c * HW]; s += v; s2 += v * v; }
  float mu  = s * (1.f / Cc);
  float var = s2 * (1.f / Cc) - mu * mu;
  float inv = rsqrtf(var + 1e-5f);
  bf16* o = Xn + (size_t)m * Cc;
  for (int c = 0; c < Cc; ++c)
    o[c] = (bf16)((xp[c * HW] - mu) * inv * g[c] + bta[c]);
}

// --- Tiled WMMA GEMM: C[M x N] = A[M x K] * W[K x N] + bias ----------------
// Block tile 64x128xBK32, 256 threads / 8 waves (4x2), 4 accumulators/wave.
constexpr int BM = 64, BN = 128, BK = 32;
constexpr int ALD = 40, BLD = 40;  // padded LDS row strides (elements)

#define GEMM_BODY(NGLOB)                                                        \
  __shared__ __align__(16) bf16 As[BM * ALD];                                   \
  __shared__ __align__(16) bf16 Bs[BN * BLD];                                   \
  int tid = threadIdx.x, lane = tid & 31, wave = tid >> 5;                      \
  int m0 = blockIdx.y * BM, n0 = blockIdx.x * BN;                               \
  int wm = wave & 3, wng = wave >> 2;                                           \
  v8f acc[4] = {{}, {}, {}, {}};                                                \
  int ar = tid >> 2, ac = (tid & 3) * 8;   /* A: 8 elems/thread */              \
  int bk = tid >> 3, bc = (tid & 7) * 16;  /* B: 16 elems/thread */             \
  void* a_lds = (void*)(As + ar * ALD + ac);                                    \
  for (int k0 = 0; k0 < Cc; k0 += BK) {                                         \
    async_copy16(a_lds, A + (size_t)(m0 + ar) * Cc + k0 + ac);                  \
    const bf16* wrow = W + (size_t)(k0 + bk) * (NGLOB) + n0 + bc;               \
    v8bf bv0 = *(const v8bf*)(wrow);                                            \
    v8bf bv1 = *(const v8bf*)(wrow + 8);                                        \
    if (k0 + BK < Cc)                                                           \
      __builtin_prefetch(wrow + (size_t)BK * (NGLOB), 0, 0);                    \
    _Pragma("unroll")                                                           \
    for (int j = 0; j < 8; ++j) {                                               \
      Bs[(bc + j) * BLD + bk]     = bv0[j];                                     \
      Bs[(bc + 8 + j) * BLD + bk] = bv1[j];                                     \
    }                                                                           \
    wait_async0();                                                              \
    __syncthreads();                                                            \
    v16bf af = frag_a(As, ALD, wm * 16 + (lane & 15), 0, lane);                 \
    _Pragma("unroll")                                                           \
    for (int ni = 0; ni < 4; ++ni) {                                            \
      v16bf bfr = frag_b(Bs, BLD, (wng * 4 + ni) * 16 + (lane & 15), 0, lane);  \
      acc[ni] = WMMA_BF16(af, bfr, acc[ni]);                                    \
    }                                                                           \
    __syncthreads();                                                            \
  }                                                                             \
  int rbase = m0 + wm * 16 + ((lane & 16) ? 8 : 0);

__global__ __launch_bounds__(256) void k_gemm_qkv(const bf16* __restrict__ A,
                                                  const bf16* __restrict__ W,
                                                  const float* __restrict__ bias,
                                                  bf16* __restrict__ Out) {
  GEMM_BODY(NN3)
#pragma unroll
  for (int ni = 0; ni < 4; ++ni) {
    int c = n0 + (wng * 4 + ni) * 16 + (lane & 15);
    float bf_ = bias[c];
#pragma unroll
    for (int r = 0; r < 8; ++r)
      Out[(size_t)(rbase + r) * NN3 + c] = (bf16)(acc[ni][r] + bf_);
  }
}

__global__ __launch_bounds__(256) void k_gemm_proj(const bf16* __restrict__ A,
                                                   const bf16* __restrict__ W,
                                                   const float* __restrict__ bias,
                                                   float* __restrict__ Out) {
  GEMM_BODY(Cc)
#pragma unroll
  for (int ni = 0; ni < 4; ++ni) {
    int c = n0 + (wng * 4 + ni) * 16 + (lane & 15);
    float bf_ = bias[c];
#pragma unroll
    for (int r = 0; r < 8; ++r) {
      int m = rbase + r, b = m >> 12, p = m & 4095;    // output is (B, C, H*W)
      Out[((size_t)b * Cc + c) * HW + p] = acc[ni][r] + bf_;
    }
  }
}

// --- Window attention: one block per (batch, branch, window, head) ---------
__global__ __launch_bounds__(128) void k_attn(const bf16* __restrict__ QKV,
                                              bf16* __restrict__ Y,
                                              const float* __restrict__ rpb1,
                                              const float* __restrict__ rpb2) {
  __shared__ __align__(16) bf16 Qs[64 * 72];  // Q (raw), K-padded to 64; reused for P
  __shared__ __align__(16) bf16 Ks[64 * 72];  // K, K-padded to 64
  __shared__ __align__(16) bf16 Vt[48 * 72];  // V transposed: [dim][token]
  __shared__ float Sf[64 * 64];               // fp32 scores / softmax scratch
  int tid = threadIdx.x, lane = tid & 31, wave = tid >> 5;
  int bid = blockIdx.x;
  int h  = bid & 3;
  int w  = (bid >> 2) & 63;
  int br = (bid >> 8) & 1;
  int b  = bid >> 9;
  int wy = w >> 3, wx = w & 7;
  int shift = br ? SS_ : 0;                   // roll(-SS) on read == roll(+SS) on write
  const float* rpb = br ? rpb2 : rpb1;

  // Stage Q/K (async 16B copies) and V^T (transpose scatter): 2 threads/token.
  {
    int t = tid >> 1, half = tid & 1;
    int ty = t >> 3, tx = t & 7;
    int y = (wy * 8 + ty + shift) & 63;
    int x = (wx * 8 + tx + shift) & 63;
    size_t m = (size_t)b * HW + y * 64 + x;
    const bf16* src = QKV + m * NN3 + br * CH + h * HD;
    v8bf z = {};
    *(v8bf*)(Qs + t * 72 + 48 + half * 8) = z;   // zero pad dims 48..63
    *(v8bf*)(Ks + t * 72 + 48 + half * 8) = z;
#pragma unroll
    for (int c = 0; c < 3; ++c) {
      int d = half * 24 + c * 8;
      async_copy16(Qs + t * 72 + d, src + d);
      async_copy16(Ks + t * 72 + d, src + Cc + d);
    }
    for (int d = half * 24; d < half * 24 + 24; ++d)
      Vt[d * 72 + t] = src[2 * Cc + d];
    wait_async0();
  }
  __syncthreads();

  // S = Q @ K^T (scale fused into epilogue): wave owns row-tile mi=wave, 4 N-tiles.
  {
    v8f acc[4] = {{}, {}, {}, {}};
    int mi = wave;
#pragma unroll
    for (int kk = 0; kk < 64; kk += 32) {
      v16bf af = frag_a(Qs, 72, mi * 16 + (lane & 15), kk, lane);
#pragma unroll
      for (int ni = 0; ni < 4; ++ni) {
        v16bf bfm = frag_b(Ks, 72, ni * 16 + (lane & 15), kk, lane);
        acc[ni] = WMMA_BF16(af, bfm, acc[ni]);
      }
    }
    // scale + relative-position bias (+ shift-window mask for branch 2)
#pragma unroll
    for (int ni = 0; ni < 4; ++ni) {
      int j = ni * 16 + (lane & 15);
      int jy = j >> 3, jx = j & 7;
      int gyj = wy * 8 + jy, gxj = wx * 8 + jx;
      int rj = (gyj < 56 ? 0 : (gyj < 60 ? 1 : 2)) * 3 + (gxj < 56 ? 0 : (gxj < 60 ? 1 : 2));
#pragma unroll
      for (int r = 0; r < 8; ++r) {
        int i = mi * 16 + r + ((lane & 16) ? 8 : 0);
        int iy = i >> 3, ix = i & 7;
        int idx = (iy - jy + 7) * 15 + (ix - jx + 7);
        float v = acc[ni][r] * SCALE_ + rpb[idx * 4 + h];
        if (br) {
          int gyi = wy * 8 + iy, gxi = wx * 8 + ix;
          int ri = (gyi < 56 ? 0 : (gyi < 60 ? 1 : 2)) * 3 + (gxi < 56 ? 0 : (gxi < 60 ? 1 : 2));
          if (ri != rj) v += -100.f;
        }
        Sf[i * 64 + j] = v;
      }
    }
  }
  __syncthreads();

  // fp32 softmax per row; probs -> bf16 into Qs (Q no longer needed).
  if (tid < 64) {
    int i = tid;
    float mx = -3.4e38f;
    for (int j = 0; j < 64; ++j) mx = fmaxf(mx, Sf[i * 64 + j]);
    float s = 0.f;
    for (int j = 0; j < 64; ++j) { float e = __expf(Sf[i * 64 + j] - mx); Sf[i * 64 + j] = e; s += e; }
    float inv = 1.f / s;
    for (int j = 0; j < 64; ++j) Qs[i * 72 + j] = (bf16)(Sf[i * 64 + j] * inv);
  }
  __syncthreads();

  // O = P @ V : wave owns row-tile mi=wave, 3 N-tiles of dim 16.
  {
    v8f acc[3] = {{}, {}, {}};
    int mi = wave;
#pragma unroll
    for (int kk = 0; kk < 64; kk += 32) {
      v16bf af = frag_a(Qs, 72, mi * 16 + (lane & 15), kk, lane);
#pragma unroll
      for (int ni = 0; ni < 3; ++ni) {
        v16bf bfm = frag_b(Vt, 72, ni * 16 + (lane & 15), kk, lane);
        acc[ni] = WMMA_BF16(af, bfm, acc[ni]);
      }
    }
#pragma unroll
    for (int ni = 0; ni < 3; ++ni) {
      int d = ni * 16 + (lane & 15);
#pragma unroll
      for (int r = 0; r < 8; ++r) {
        int i = mi * 16 + r + ((lane & 16) ? 8 : 0);
        int ty = i >> 3, tx = i & 7;
        int y = (wy * 8 + ty + shift) & 63;
        int x = (wx * 8 + tx + shift) & 63;
        size_t m = (size_t)b * HW + y * 64 + x;
        Y[m * Cc + br * CH + h * HD + d] = (bf16)acc[ni][r];
      }
    }
  }
}

// ---------------------------------------------------------------------------
extern "C" void kernel_launch(void* const* d_in, const int* in_sizes, int n_in,
                              void* d_out, int out_size, void* d_ws, size_t ws_size,
                              hipStream_t stream) {
  const float* x      = (const float*)d_in[0];
  const float* norm_g = (const float*)d_in[1];
  const float* norm_b = (const float*)d_in[2];
  const float* qkv_w  = (const float*)d_in[3];
  const float* qkv_b  = (const float*)d_in[4];
  const float* proj_w = (const float*)d_in[5];
  const float* proj_b = (const float*)d_in[6];
  const float* rpb1   = (const float*)d_in[7];
  const float* rpb2   = (const float*)d_in[8];
  float* out = (float*)d_out;

  char* ws = (char*)d_ws;
  bf16* wq  = (bf16*)(ws);                       // 384*1152*2      =    884,736
  bf16* wp  = (bf16*)(ws + 884736);              // 384*384*2       =    294,912
  bf16* Xn  = (bf16*)(ws + 1179648);             // 32768*384*2     = 25,165,824
  bf16* QKV = (bf16*)(ws + 26345472);            // 32768*1152*2    = 75,497,472
  bf16* Yb  = (bf16*)(ws + 101842944);           // 32768*384*2     = 25,165,824

  int ncvt = Cc * NN3 + Cc * Cc;
  k_cvt<<<(ncvt + 255) / 256, 256, 0, stream>>>(qkv_w, proj_w, wq, wp);
  k_ln<<<Mtok / 128, 128, 0, stream>>>(x, norm_g, norm_b, Xn);
  dim3 g1(NN3 / BN, Mtok / BM);                  // (9, 512)
  k_gemm_qkv<<<g1, 256, 0, stream>>>(Xn, wq, qkv_b, QKV);
  k_attn<<<Bn * 2 * 64 * 4, 128, 0, stream>>>(QKV, Yb, rpb1, rpb2);
  dim3 g2(Cc / BN, Mtok / BM);                   // (3, 512)
  k_gemm_proj<<<g2, 256, 0, stream>>>(Yb, wp, proj_b, out);
}
